// RNN_86870008528979
// MI455X (gfx1250) — matrix operand order repaired and checked by reading, not compiled, not run on hardware
//
#include <hip/hip_runtime.h>
#include <math.h>
#include <stdint.h>

// ---------------------------------------------------------------------------
// Stepped 2-layer RNN (tanh) + linear readout, bf16 WMMA on gfx1250 (wave32).
//   h1 = tanh([x, h1p] @ W1 + b1)      K = 512 + 1536
//   h2 = tanh([h1, h2p] @ W2 + b2)     K = 1536 + 1536
//   out = h2 @ Wo + bo                 K = 1536
// Weights (35MB fp32 -> 17MB bf16) are L2-resident (192MB L2); the scan is
// serial, so each step is 3 graph-captured kernels (deterministic sync).
// Occupancy is ~1 wave/SIMD, so the K-loops are software-pipelined in
// registers: next K-step's A/B tiles are loaded before the current WMMAs.
// ---------------------------------------------------------------------------

typedef __bf16 bf16_t;
typedef __attribute__((ext_vector_type(16))) __bf16 v16bf;
typedef __attribute__((ext_vector_type(8)))  float  v8f;
typedef __attribute__((ext_vector_type(4)))  float  v4f;

#define RNN_STEPS 256
#define RNN_B     128
#define RNN_IN    512
#define RNN_H1    1536
#define RNN_H2    1536
#define RNN_OUT   512

// ---- tile loaders ---------------------------------------------------------
// A-matrix 16x32 bf16, this code's contiguous k order (shared with B): lane<16 holds K 0..15 of row
// (lane&15); lane>=16 holds K 16..31. Two 16B loads per lane.
__device__ __forceinline__ v16bf load_a(const bf16_t* __restrict__ p) {
    union { v4f f[2]; v16bf v; } u;
    u.f[0] = *reinterpret_cast<const v4f*>(p);
    u.f[1] = *reinterpret_cast<const v4f*>(p + 8);
    return u.v;
}
// Same A layout but fp32 source (raw input x): convert on load.
__device__ __forceinline__ v16bf load_a(const float* __restrict__ p) {
    v4f a = *reinterpret_cast<const v4f*>(p);
    v4f b = *reinterpret_cast<const v4f*>(p + 4);
    v4f c = *reinterpret_cast<const v4f*>(p + 8);
    v4f d = *reinterpret_cast<const v4f*>(p + 12);
    v16bf r;
#pragma unroll
    for (int i = 0; i < 4; ++i) {
        r[i]      = (bf16_t)a[i];
        r[i + 4]  = (bf16_t)b[i];
        r[i + 8]  = (bf16_t)c[i];
        r[i + 12] = (bf16_t)d[i];
    }
    return r;
}
// B-matrix 32x16 bf16, same contiguous k order as A: lane = column (lane&15); lanes 0-15 hold K=0..15,
// lanes 16-31 hold K=16..31 -> with W stored transposed (NxK row-major) this
// is one contiguous 32B load per lane.
__device__ __forceinline__ v16bf load_b(const bf16_t* __restrict__ p) {
    union { v4f f[2]; v16bf v; } u;
    u.f[0] = reinterpret_cast<const v4f*>(p)[0];
    u.f[1] = reinterpret_cast<const v4f*>(p)[1];
    return u.v;
}

#define WMMA_BF16(A, B, C) \
    __builtin_amdgcn_wmma_f32_16x16x32_bf16(false, (A), false, (B), (short)0, (C), false, false)

// ---- software-pipelined K-partial GEMM (one 16x64 strip per wave) ---------
// arowp: this lane's A row pointer (already + ah). wrow: this lane's B row
// pointer for n-tile j=0 (already + part K-offset + bh); j stride = 16*KT.
template <int K, int KT, typename SRC>
__device__ __forceinline__ void gemm_part(const SRC* __restrict__ arowp,
                                          const bf16_t* __restrict__ wrow,
                                          v8f acc[4]) {
    constexpr size_t JS = (size_t)16 * KT;
    v16bf a  = load_a(arowp);
    v16bf b0 = load_b(wrow);
    v16bf b1 = load_b(wrow + JS);
    v16bf b2 = load_b(wrow + 2 * JS);
    v16bf b3 = load_b(wrow + 3 * JS);
#pragma unroll 2
    for (int k = 32; k < K; k += 32) {
        // prefetch next K-step into registers before using current tiles
        v16bf an = load_a(arowp + k);
        v16bf c0 = load_b(wrow + k);
        v16bf c1 = load_b(wrow + JS + k);
        v16bf c2 = load_b(wrow + 2 * JS + k);
        v16bf c3 = load_b(wrow + 3 * JS + k);
        acc[0] = WMMA_BF16(a, b0, acc[0]);
        acc[1] = WMMA_BF16(a, b1, acc[1]);
        acc[2] = WMMA_BF16(a, b2, acc[2]);
        acc[3] = WMMA_BF16(a, b3, acc[3]);
        a = an; b0 = c0; b1 = c1; b2 = c2; b3 = c3;
    }
    acc[0] = WMMA_BF16(a, b0, acc[0]);
    acc[1] = WMMA_BF16(a, b1, acc[1]);
    acc[2] = WMMA_BF16(a, b2, acc[2]);
    acc[3] = WMMA_BF16(a, b3, acc[3]);
}

// ---- one RNN layer: D = act([A0|A1|A2] @ Wt^T + bias) ---------------------
// Compile-time shapes so K-loops have constant trip counts and wave->tile
// mapping is mul/shift. Each wave owns a 16x64 output strip; grids are exact.
template <int K0, int K1, int K2, int N, bool TANH>
__global__ __launch_bounds__(256) void rnn_layer(
    const float*  __restrict__ A0,
    const bf16_t* __restrict__ A1,
    const bf16_t* __restrict__ A2,
    const bf16_t* __restrict__ Wt,   // N x (K0+K1+K2), bf16, row-major
    const float*  __restrict__ bias,
    bf16_t* __restrict__ outb,       // TANH:  bf16 hidden state
    float*  __restrict__ outf)       // !TANH: fp32 readout
{
    constexpr int KT  = K0 + K1 + K2;
    constexpr int NWT = N / 64;
    const int lane = threadIdx.x & 31;
    const int wave = blockIdx.x * 8 + (threadIdx.x >> 5);
    const int mt = wave / NWT;
    const int nt = wave - mt * NWT;
    const int m0 = mt << 4;
    const int n0 = nt << 6;
    const int arow = lane & 15;
    const int ah   = (lane >> 4) << 4;   // A K-half select: 0 or 16
    const int bh   = (lane >> 4) << 4;   // B K-half select: 0 or 16

    const bf16_t* wrow = Wt + (size_t)(n0 + arow) * KT + bh;
    v8f acc[4] = {v8f{}, v8f{}, v8f{}, v8f{}};

    if constexpr (K0 > 0)
        gemm_part<K0, KT>(A0 + (size_t)(m0 + arow) * K0 + ah, wrow, acc);
    if constexpr (K1 > 0)
        gemm_part<K1, KT>(A1 + (size_t)(m0 + arow) * K1 + ah, wrow + K0, acc);
    if constexpr (K2 > 0)
        gemm_part<K2, KT>(A2 + (size_t)(m0 + arow) * K2 + ah, wrow + K0 + K1, acc);

    // C/D layout: VGPR r -> row m0 + (lane>=16)*8 + r, col n0 + (lane&15)
    const int crow = (lane >> 4) << 3;
#pragma unroll
    for (int j = 0; j < 4; ++j) {
        const int n  = n0 + j * 16 + arow;
        const float bv = bias[n];
#pragma unroll
        for (int r = 0; r < 8; ++r) {
            float v = acc[j][r] + bv;
            const size_t idx = (size_t)(m0 + crow + r) * N + n;
            if constexpr (TANH) outb[idx] = (bf16_t)tanhf(v);
            else                outf[idx] = v;
        }
    }
}

// ---- helpers: weight transpose+convert, zero fill -------------------------
__global__ void convert_wt(const float* __restrict__ W, bf16_t* __restrict__ Wt,
                           int K, int N) {
    int idx = blockIdx.x * blockDim.x + threadIdx.x;
    if (idx >= K * N) return;
    int n = idx / K, k = idx - n * K;          // Wt[n][k] = W[k][n]
    Wt[idx] = (bf16_t)W[(size_t)k * N + n];    // contiguous writes
}

__global__ void zero_u32(uint32_t* __restrict__ p, int n) {
    int i = blockIdx.x * blockDim.x + threadIdx.x;
    if (i < n) p[i] = 0u;
}

// ---------------------------------------------------------------------------
extern "C" void kernel_launch(void* const* d_in, const int* in_sizes, int n_in,
                              void* d_out, int out_size, void* d_ws, size_t ws_size,
                              hipStream_t stream) {
    const float* inputs = (const float*)d_in[0];   // [256,128,512]
    const float* W1     = (const float*)d_in[1];   // [2048,1536]
    const float* b1     = (const float*)d_in[2];   // [1536]
    const float* W2     = (const float*)d_in[3];   // [3072,1536]
    const float* b2     = (const float*)d_in[4];   // [1536]
    const float* Wo     = (const float*)d_in[5];   // [1536,512]
    const float* bo     = (const float*)d_in[6];   // [512]
    float* out = (float*)d_out;                    // [257,128,512]

    const int K1tot = RNN_IN + RNN_H1;   // 2048
    const int K2tot = RNN_H1 + RNN_H2;   // 3072

    // workspace carve-up (~18.9 MB)
    char* ws = (char*)d_ws;
    bf16_t* w1t = (bf16_t*)ws; ws += (size_t)RNN_H1 * K1tot * sizeof(bf16_t);
    bf16_t* w2t = (bf16_t*)ws; ws += (size_t)RNN_H2 * K2tot * sizeof(bf16_t);
    bf16_t* wot = (bf16_t*)ws; ws += (size_t)RNN_OUT * RNN_H2 * sizeof(bf16_t);
    bf16_t* h1buf[2]; bf16_t* h2buf[2];
    for (int i = 0; i < 2; ++i) { h1buf[i] = (bf16_t*)ws; ws += (size_t)RNN_B * RNN_H1 * sizeof(bf16_t); }
    for (int i = 0; i < 2; ++i) { h2buf[i] = (bf16_t*)ws; ws += (size_t)RNN_B * RNN_H2 * sizeof(bf16_t); }

    // weight transpose + fp32->bf16 (re-done every call; deterministic)
    {
        int t1 = K1tot * RNN_H1, t2 = K2tot * RNN_H2, t3 = RNN_H2 * RNN_OUT;
        convert_wt<<<(t1 + 255) / 256, 256, 0, stream>>>(W1, w1t, K1tot, RNN_H1);
        convert_wt<<<(t2 + 255) / 256, 256, 0, stream>>>(W2, w2t, K2tot, RNN_H2);
        convert_wt<<<(t3 + 255) / 256, 256, 0, stream>>>(Wo, wot, RNN_H2, RNN_OUT);
    }
    // zero initial hidden states + delay slot of the output
    {
        int nh = RNN_B * RNN_H1 * (int)sizeof(bf16_t) / 4;
        zero_u32<<<(nh + 255) / 256, 256, 0, stream>>>((uint32_t*)h1buf[0], nh);
        zero_u32<<<(nh + 255) / 256, 256, 0, stream>>>((uint32_t*)h2buf[0], nh);
        int no = RNN_B * RNN_OUT;
        zero_u32<<<(no + 255) / 256, 256, 0, stream>>>((uint32_t*)out, no);
    }

    // grid: waves = (B/16) * (N/64); 8 waves (256 thr) per block (exact fit)
    const int blks_h   = (RNN_B / 16) * (RNN_H1 / 64) / 8;   // 24
    const int blks_out = (RNN_B / 16) * (RNN_OUT / 64) / 8;  // 8

    int p = 0;
    for (int t = 0; t < RNN_STEPS; ++t) {
        const float* x_t = inputs + (size_t)t * RNN_B * RNN_IN;
        // hidden1 = tanh([x_t, h1_prev] @ W1 + b1)
        rnn_layer<RNN_IN, RNN_H1, 0, RNN_H1, true><<<blks_h, 256, 0, stream>>>(
            x_t, h1buf[p], nullptr, w1t, b1, h1buf[1 - p], nullptr);
        // hidden2 = tanh([h1_cur, h2_prev] @ W2 + b2)
        rnn_layer<0, RNN_H1, RNN_H2, RNN_H2, true><<<blks_h, 256, 0, stream>>>(
            nullptr, h1buf[1 - p], h2buf[p], w2t, b2, h2buf[1 - p], nullptr);
        // out[t+1] = h2_cur @ Wo + bo
        rnn_layer<0, RNN_H2, 0, RNN_OUT, false><<<blks_out, 256, 0, stream>>>(
            nullptr, h2buf[1 - p], nullptr, wot, bo, nullptr,
            out + (size_t)(t + 1) * RNN_B * RNN_OUT);
        p ^= 1;
    }
}
